// Block_69312182223602
// MI455X (gfx1250) — matrix operand order repaired and checked by reading, not compiled, not run on hardware
//
#include <hip/hip_runtime.h>
#include <hip/hip_bf16.h>
#include <math.h>

#define B_  8
#define T_  1024
#define C_  768
#define H_  12
#define HD_ 64
#define M_  (B_ * T_)

typedef __attribute__((ext_vector_type(16))) __bf16 v16bf;
typedef __attribute__((ext_vector_type(8)))  float  v8f;

union FragLd { uint4 q[2]; v16bf v; unsigned short u[16]; };

__device__ __forceinline__ unsigned short f2bf(float x) {
  unsigned int u = __float_as_uint(x);
  u += 0x7fffu + ((u >> 16) & 1u);          // round-to-nearest-even
  return (unsigned short)(u >> 16);
}

__device__ __forceinline__ v8f bwmma(v16bf a, v16bf b, v8f c) {
  return __builtin_amdgcn_wmma_f32_16x16x32_bf16(false, a, false, b,
                                                 (short)0, c, false, false);
}

// ---- CDNA5 async global->LDS copy (ASYNCcnt path), 16B per lane ----
__device__ __forceinline__ void async_load_b128(unsigned lds_off, const void* gptr) {
  unsigned long long ga = (unsigned long long)(size_t)gptr;
  asm volatile("global_load_async_to_lds_b128 %0, %1, off"
               :: "v"(lds_off), "v"(ga) : "memory");
}
__device__ __forceinline__ void wait_async0() {
  asm volatile("s_wait_asynccnt 0" ::: "memory");
}

// A-operand fragment (16x32 bf16) from a row-major tile.
// Per-lane halves are two contiguous 16B runs -> 2x ds_load_b128.
__device__ __forceinline__ v16bf load_frag_row(const unsigned short* p, int stride) {
  int lane = threadIdx.x & 31;
  const unsigned short* row = p + (lane & 15) * stride + ((lane >> 4) & 1) * 8;
  FragLd f;
  f.q[0] = *(const uint4*)(row);
  f.q[1] = *(const uint4*)(row + 16);
  return f.v;
}

// B-operand fragment (32x16 bf16) from a tile stored N-major: tile[n][k].
// Per-lane halves are two contiguous 16B runs -> 2x ds_load_b128.
__device__ __forceinline__ v16bf load_frag_bn(const unsigned short* p, int stride) {
  int lane = threadIdx.x & 31;
  const unsigned short* row = p + (lane & 15) * stride + ((lane >> 4) & 1) * 8;
  FragLd f;
  f.q[0] = *(const uint4*)(row);
  f.q[1] = *(const uint4*)(row + 16);
  return f.v;
}

__device__ __forceinline__ float gelu_f(float x) {
  float x3 = x * x * x;
  return 0.5f * x * (1.0f + tanhf(0.7978845608028654f * (x + 0.044715f * x3)));
}

// -------- f32 [K,N] -> bf16 transposed [N,K] (weights, one-time) --------
__launch_bounds__(256)
__global__ void cvtT_kernel(const float* __restrict__ in,
                            unsigned short* __restrict__ out, int K, int N) {
  __shared__ unsigned short tile[32][33];
  int k0 = blockIdx.y * 32, n0 = blockIdx.x * 32;
  int tx = threadIdx.x & 31, ty = threadIdx.x >> 5;
#pragma unroll
  for (int i = ty; i < 32; i += 8)
    tile[i][tx] = f2bf(in[(size_t)(k0 + i) * N + n0 + tx]);
  __syncthreads();
#pragma unroll
  for (int i = ty; i < 32; i += 8)
    out[(size_t)(n0 + i) * K + k0 + tx] = tile[tx][i];
}

// -------- LayerNorm: one row of C=768 per block --------
__launch_bounds__(256)
__global__ void ln_kernel(const float* __restrict__ x,
                          const float* __restrict__ sc,
                          const float* __restrict__ bi,
                          unsigned short* __restrict__ out) {
  int row = blockIdx.x;
  int t = threadIdx.x;
  const float* xr = x + (size_t)row * C_;
  float v0 = xr[t], v1 = xr[t + 256], v2 = xr[t + 512];
  __shared__ float rs[256], rq[256];
  rs[t] = v0 + v1 + v2;
  rq[t] = v0 * v0 + v1 * v1 + v2 * v2;
  __syncthreads();
  for (int off = 128; off > 0; off >>= 1) {
    if (t < off) { rs[t] += rs[t + off]; rq[t] += rq[t + off]; }
    __syncthreads();
  }
  float mu  = rs[0] * (1.0f / C_);
  float var = rq[0] * (1.0f / C_) - mu * mu;
  float inv = rsqrtf(var + 1e-5f);
  unsigned short* orow = out + (size_t)row * C_;
  orow[t]       = f2bf((v0 - mu) * inv * sc[t]       + bi[t]);
  orow[t + 256] = f2bf((v1 - mu) * inv * sc[t + 256] + bi[t + 256]);
  orow[t + 512] = f2bf((v2 - mu) * inv * sc[t + 512] + bi[t + 512]);
}

// -------- tiled bf16 GEMM: out = A[M,K] @ W[K,N] + bias, W given as WT[N,K] --------
// 256 thr = 8 waves, 64x64 tile, BK=32, double-buffered LDS fed by async copies.
__launch_bounds__(256)
__global__ void gemm_bf16_kernel(const unsigned short* __restrict__ A,
                                 const unsigned short* __restrict__ WT,
                                 const float* __restrict__ bias,
                                 const float* __restrict__ resid,
                                 float* __restrict__ outF,
                                 unsigned short* __restrict__ outB,
                                 int M, int N, int K, int gelu) {
  __shared__ __attribute__((aligned(16))) unsigned short As[2][64][40];
  __shared__ __attribute__((aligned(16))) unsigned short Bs[2][64][40];
  const int bm = blockIdx.y * 64, bn = blockIdx.x * 64;
  const int tid = threadIdx.x, wave = tid >> 5, lane = tid & 31;
  const int wm = (wave & 3) * 16, wn = (wave >> 2) * 32;
  const int r = tid >> 2, c = (tid & 3) * 8;      // 8 halves (16B) per thread
  const unsigned short* agp = A  + (size_t)(bm + r) * K + c;
  const unsigned short* bgp = WT + (size_t)(bn + r) * K + c;

  async_load_b128((unsigned)(size_t)&As[0][r][c], agp);
  async_load_b128((unsigned)(size_t)&Bs[0][r][c], bgp);
  v8f acc0 = {}, acc1 = {};
  const int nk = K >> 5;
  int cur = 0;
  wait_async0();
  __syncthreads();
  for (int kt = 0; kt < nk; ++kt) {
    if (kt + 1 < nk) {      // stream next tile into the other buffer
      async_load_b128((unsigned)(size_t)&As[cur ^ 1][r][c], agp + (kt + 1) * 32);
      async_load_b128((unsigned)(size_t)&Bs[cur ^ 1][r][c], bgp + (kt + 1) * 32);
      if (kt + 2 < nk) {
        __builtin_prefetch(agp + (kt + 2) * 32, 0, 3);
        __builtin_prefetch(bgp + (kt + 2) * 32, 0, 3);
      }
    }
    v16bf af = load_frag_row(&As[cur][wm][0], 40);
    v16bf b0 = load_frag_bn(&Bs[cur][wn][0], 40);
    v16bf b1 = load_frag_bn(&Bs[cur][wn + 16][0], 40);
    acc0 = bwmma(af, b0, acc0);
    acc1 = bwmma(af, b1, acc1);
    wait_async0();
    __syncthreads();
    cur ^= 1;
  }
  const int n0 = bn + wn + (lane & 15);
  const float bias0 = bias[n0], bias1 = bias[n0 + 16];
#pragma unroll
  for (int i = 0; i < 8; ++i) {
    int row = bm + wm + ((lane >> 4) << 3) + i;
    float v0 = acc0[i] + bias0;
    float v1 = acc1[i] + bias1;
    if (gelu) { v0 = gelu_f(v0); v1 = gelu_f(v1); }
    if (resid) {
      v0 += resid[(size_t)row * N + n0];
      v1 += resid[(size_t)row * N + n0 + 16];
    }
    if (outF) {
      outF[(size_t)row * N + n0]      = v0;
      outF[(size_t)row * N + n0 + 16] = v1;
    }
    if (outB) {
      outB[(size_t)row * N + n0]      = f2bf(v0);
      outB[(size_t)row * N + n0 + 16] = f2bf(v1);
    }
  }
}

// -------- causal flash attention --------
// grid = B*H*(T/64); 128 thr = 4 waves; wave owns 16 query rows, KV tiles of 32
__launch_bounds__(128)
__global__ void attn_kernel(const unsigned short* __restrict__ qkv,
                            unsigned short* __restrict__ ctx) {
  __shared__ __attribute__((aligned(16))) unsigned short Ks[32][72];   // [kv][d]
  __shared__ __attribute__((aligned(16))) unsigned short VsT[64][40];  // [d][kv]
  __shared__ __attribute__((aligned(16))) unsigned short Ps[4][16][40];
  const int nqb = T_ / 64;
  int idx = blockIdx.x;
  int qb  = (idx % nqb) * 64;
  int bh  = idx / nqb;
  int h = bh % H_, b = bh / H_;
  const int tid = threadIdx.x, wave = tid >> 5, lane = tid & 31;
  const int qr = qb + wave * 16;
  const size_t rstr = 3 * C_;

  // Q fragments (K=0..31, 32..63): contiguous 16B runs -> 4x global_load_b128
  FragLd qf0, qf1;
  {
    int m = lane & 15, kh = ((lane >> 4) & 1) * 8;
    const unsigned short* qrow =
        qkv + (size_t)(b * T_ + qr + m) * rstr + h * HD_ + kh;
    qf0.q[0] = *(const uint4*)(qrow);
    qf0.q[1] = *(const uint4*)(qrow + 16);
    qf1.q[0] = *(const uint4*)(qrow + 32);
    qf1.q[1] = *(const uint4*)(qrow + 48);
  }

  v8f o0 = {}, o1 = {}, o2 = {}, o3 = {};
  float mrow[8], lrow[8];
#pragma unroll
  for (int i = 0; i < 8; ++i) { mrow[i] = -__builtin_inff(); lrow[i] = 0.0f; }

  const int lr = tid >> 2;            // kv row this thread handles
  const int lc = (tid & 3) * 16;      // d offset
  const int ktmax = (qb + 63) >> 5;
  for (int kt = 0; kt <= ktmax; ++kt) {
    int kbase = kt * 32;
    const unsigned short* ksrc =
        qkv + (size_t)(b * T_ + kbase + lr) * rstr + C_ + h * HD_ + lc;
    const unsigned short* vsrc = ksrc + C_;
    FragLd vv;
    vv.q[0] = *(const uint4*)(vsrc);
    vv.q[1] = *(const uint4*)(vsrc + 8);
    __syncthreads();                       // prior tile fully consumed
    async_load_b128((unsigned)(size_t)&Ks[lr][lc], ksrc);       // K tile: async
    async_load_b128((unsigned)(size_t)&Ks[lr][lc + 8], ksrc + 8);
#pragma unroll
    for (int j = 0; j < 16; ++j)           // V tile: transpose into [d][kv]
      VsT[lc + j][lr] = vv.u[j];
    wait_async0();
    __syncthreads();

    // S = Q K^T (two 16x16 score tiles, K=64 = 2 WMMA steps each)
    v8f s0 = {}, s1 = {};
    s0 = bwmma(qf0.v, load_frag_bn(&Ks[0][0],   72), s0);
    s0 = bwmma(qf1.v, load_frag_bn(&Ks[0][32],  72), s0);
    s1 = bwmma(qf0.v, load_frag_bn(&Ks[16][0],  72), s1);
    s1 = bwmma(qf1.v, load_frag_bn(&Ks[16][32], 72), s1);

    int kg = kbase + (lane & 15);
#pragma unroll
    for (int i = 0; i < 8; ++i) {
      int rl = ((lane >> 4) << 3) + i;
      int qg = qr + rl;
      float e0 = (kg      <= qg) ? s0[i] * 0.125f : -__builtin_inff();
      float e1 = (kg + 16 <= qg) ? s1[i] * 0.125f : -__builtin_inff();
      float rm = fmaxf(e0, e1);
      rm = fmaxf(rm, __shfl_xor(rm, 1, 32));
      rm = fmaxf(rm, __shfl_xor(rm, 2, 32));
      rm = fmaxf(rm, __shfl_xor(rm, 4, 32));
      rm = fmaxf(rm, __shfl_xor(rm, 8, 32));
      float newm  = fmaxf(mrow[i], rm);
      float alpha = __expf(mrow[i] - newm);
      float p0 = __expf(e0 - newm);
      float p1 = __expf(e1 - newm);
      float rs = p0 + p1;
      rs += __shfl_xor(rs, 1, 32);
      rs += __shfl_xor(rs, 2, 32);
      rs += __shfl_xor(rs, 4, 32);
      rs += __shfl_xor(rs, 8, 32);
      lrow[i] = lrow[i] * alpha + rs;
      mrow[i] = newm;
      o0[i] *= alpha; o1[i] *= alpha; o2[i] *= alpha; o3[i] *= alpha;
      Ps[wave][rl][lane & 15]        = f2bf(p0);
      Ps[wave][rl][(lane & 15) + 16] = f2bf(p1);
    }
    // O += P @ V  (per-wave LDS region; same-wave DS ops are in-order)
    v16bf pf = load_frag_row(&Ps[wave][0][0], 40);
    o0 = bwmma(pf, load_frag_bn(&VsT[0][0],  40), o0);
    o1 = bwmma(pf, load_frag_bn(&VsT[16][0], 40), o1);
    o2 = bwmma(pf, load_frag_bn(&VsT[32][0], 40), o2);
    o3 = bwmma(pf, load_frag_bn(&VsT[48][0], 40), o3);
  }

  int n = lane & 15;
#pragma unroll
  for (int i = 0; i < 8; ++i) {
    int rl = ((lane >> 4) << 3) + i;
    float inv = 1.0f / lrow[i];
    size_t base = (size_t)(b * T_ + qr + rl) * C_ + h * HD_ + n;
    ctx[base]      = f2bf(o0[i] * inv);
    ctx[base + 16] = f2bf(o1[i] * inv);
    ctx[base + 32] = f2bf(o2[i] * inv);
    ctx[base + 48] = f2bf(o3[i] * inv);
  }
}

extern "C" void kernel_launch(void* const* d_in, const int* in_sizes, int n_in,
                              void* d_out, int out_size, void* d_ws, size_t ws_size,
                              hipStream_t stream) {
  const float* x    = (const float*)d_in[0];
  // d_in[1] = mask (unused; causal mask computed analytically)
  const float* ln1s = (const float*)d_in[2];
  const float* ln1b = (const float*)d_in[3];
  const float* wqkv = (const float*)d_in[4];
  const float* bqkv = (const float*)d_in[5];
  const float* wap  = (const float*)d_in[6];
  const float* bap  = (const float*)d_in[7];
  const float* ln2s = (const float*)d_in[8];
  const float* ln2b = (const float*)d_in[9];
  const float* wfc  = (const float*)d_in[10];
  const float* bfc  = (const float*)d_in[11];
  const float* wmp  = (const float*)d_in[12];
  const float* bmp  = (const float*)d_in[13];
  float* out = (float*)d_out;

  char* ws = (char*)d_ws;
  size_t off = 0;
  auto alloc = [&](size_t bytes) -> void* {
    void* p = ws + off;
    off = (off + bytes + 255) & ~(size_t)255;
    return p;
  };
  unsigned short* wqkv_t = (unsigned short*)alloc((size_t)C_ * 3 * C_ * 2);
  unsigned short* wap_t  = (unsigned short*)alloc((size_t)C_ * C_ * 2);
  unsigned short* wfc_t  = (unsigned short*)alloc((size_t)C_ * 4 * C_ * 2);
  unsigned short* wmp_t  = (unsigned short*)alloc((size_t)4 * C_ * C_ * 2);
  unsigned short* h1     = (unsigned short*)alloc((size_t)M_ * C_ * 2);
  unsigned short* qkvb   = (unsigned short*)alloc((size_t)M_ * 3 * C_ * 2);
  unsigned short* ctxb   = (unsigned short*)alloc((size_t)M_ * C_ * 2);
  float*          x1     = (float*)alloc((size_t)M_ * C_ * 4);
  unsigned short* h2     = (unsigned short*)alloc((size_t)M_ * C_ * 2);
  unsigned short* actb   = (unsigned short*)alloc((size_t)M_ * 4 * C_ * 2);
  (void)ws_size; (void)n_in; (void)in_sizes; (void)out_size;

  // 1) weights f32 -> bf16, transposed to [N][K] for K-major fragment loads
  { dim3 g(3 * C_ / 32, C_ / 32);
    cvtT_kernel<<<g, 256, 0, stream>>>(wqkv, wqkv_t, C_, 3 * C_); }
  { dim3 g(C_ / 32, C_ / 32);
    cvtT_kernel<<<g, 256, 0, stream>>>(wap, wap_t, C_, C_); }
  { dim3 g(4 * C_ / 32, C_ / 32);
    cvtT_kernel<<<g, 256, 0, stream>>>(wfc, wfc_t, C_, 4 * C_); }
  { dim3 g(C_ / 32, 4 * C_ / 32);
    cvtT_kernel<<<g, 256, 0, stream>>>(wmp, wmp_t, 4 * C_, C_); }

  // 2) LN1
  ln_kernel<<<M_, 256, 0, stream>>>(x, ln1s, ln1b, h1);

  // 3) QKV projection: [8192,768] @ [768,2304]
  { dim3 g(3 * C_ / 64, M_ / 64);
    gemm_bf16_kernel<<<g, 256, 0, stream>>>(h1, wqkv_t, bqkv, nullptr,
                                            nullptr, qkvb, M_, 3 * C_, C_, 0); }

  // 4) causal flash attention -> ctx
  attn_kernel<<<B_ * H_ * (T_ / 64), 128, 0, stream>>>(qkvb, ctxb);

  // 5) attn proj + residual: x1 = x + ctx @ w_attn_proj + b
  { dim3 g(C_ / 64, M_ / 64);
    gemm_bf16_kernel<<<g, 256, 0, stream>>>(ctxb, wap_t, bap, x,
                                            x1, nullptr, M_, C_, C_, 0); }

  // 6) LN2
  ln_kernel<<<M_, 256, 0, stream>>>(x1, ln2s, ln2b, h2);

  // 7) FC + GELU: [8192,768] @ [768,3072]
  { dim3 g(4 * C_ / 64, M_ / 64);
    gemm_bf16_kernel<<<g, 256, 0, stream>>>(h2, wfc_t, bfc, nullptr,
                                            nullptr, actb, M_, 4 * C_, C_, 1); }

  // 8) MLP proj + residual -> d_out (f32)
  { dim3 g(C_ / 64, M_ / 64);
    gemm_bf16_kernel<<<g, 256, 0, stream>>>(actb, wmp_t, bmp, x1,
                                            out, nullptr, M_, C_, 4 * C_, 0); }
}
